// ScaledDotProductAttention_28174985462575
// MI455X (gfx1250) — matrix operand order repaired and checked by reading, not compiled, not run on hardware
//
#include <hip/hip_runtime.h>
#include <stdint.h>

typedef __attribute__((ext_vector_type(16))) _Float16 v16h;
typedef __attribute__((ext_vector_type(8)))  _Float16 v8h;
typedef __attribute__((ext_vector_type(4)))  _Float16 v4h;
typedef __attribute__((ext_vector_type(8)))  float    v8f;
typedef __attribute__((ext_vector_type(4)))  float    f32x4;
typedef unsigned int u32x4 __attribute__((ext_vector_type(4)));
typedef int          i32x4 __attribute__((ext_vector_type(4)));
typedef int          i32x8 __attribute__((ext_vector_type(8)));

#define SEQ  4096
#define HD   64
#define NH   16
#define CHK  64                       /* k-columns per staged chunk */
#define NCH  (SEQ / CHK)
#define L2E  1.4426950408889634f
#define SCL2 (0.125f * L2E)           /* 1/sqrt(64) * log2(e), folded into Q */
#define NEG2 (-10000.0f * L2E)

__device__ __forceinline__ v8f wmma32f16(v16h a, v16h b, v8f c) {
  return __builtin_amdgcn_wmma_f32_16x16x32_f16(false, a, false, b, (short)0, c, false, false);
}

__device__ __forceinline__ v16h ld16h(const _Float16* p) {
  v8h lo = *(const v8h*)(p);
  v8h hi = *(const v8h*)(p + 8);
  return __builtin_shufflevector(lo, hi, 0,1,2,3,4,5,6,7,8,9,10,11,12,13,14,15);
}

// ---- Tensor Data Mover: DMA one dense 2D f16 tile (tile_x x tile_y) from global to LDS ----
__device__ __forceinline__ void tdm_load_2d(uint32_t lds_addr, const _Float16* gptr,
                                            uint32_t tile_x, uint32_t tile_y,
                                            uint32_t row_stride_elems) {
  uint64_t ga = (uint64_t)(uintptr_t)gptr;
  u32x4 g0;
  g0[0] = 1u;                                              // count=1, user descriptor
  g0[1] = lds_addr;                                        // LDS byte address
  g0[2] = (uint32_t)ga;                                    // global_addr[31:0]
  g0[3] = (uint32_t)((ga >> 32) & 0x1FFFFFFu) | (2u << 30);// global_addr[56:32] | type=2
  i32x8 g1;
  g1[0] = (int)(1u << 16);                                 // workgroup_mask=0, data_size=2B
  g1[1] = 0;                                               // tensor_dim0[15:0] (dim0 = 1<<20)
  g1[2] = (int)0x10u;                                      // tensor_dim0[31:16], dim1[15:0]=0
  g1[3] = (int)(0x10u | (tile_x << 16));                   // tensor_dim1[31:16], tile_dim0
  g1[4] = (int)tile_y;                                     // tile_dim1 (tile_dim2=0)
  g1[5] = (int)row_stride_elems;                           // tensor_dim0_stride[31:0]
  g1[6] = 0;
  g1[7] = 0;
  i32x4 z4 = {0, 0, 0, 0};
#if __clang_major__ >= 23
  i32x8 z8 = {0, 0, 0, 0, 0, 0, 0, 0};
  __builtin_amdgcn_tensor_load_to_lds(g0, g1, z4, z4, z8, 0);
#else
  __builtin_amdgcn_tensor_load_to_lds(g0, g1, z4, z4, 0);
#endif
}

// ================= pre-kernel 1: q-transposed mask bitpack (64MB int32 -> 2MB bits) ========
__global__ __launch_bounds__(256)
void pack_mask_kernel(const int* __restrict__ mask, uint32_t* __restrict__ bits) {
  int kcol = blockIdx.x * 256 + threadIdx.x;
  int qw   = blockIdx.y;
  uint32_t w = 0;
  #pragma unroll 4
  for (int i = 0; i < 32; ++i)
    w |= (mask[(size_t)(qw * 32 + i) * SEQ + kcol] != 0) ? (1u << i) : 0u;
  bits[(size_t)qw * SEQ + kcol] = w;
}

// ========== pre-kernel 2: K -> f16 row-major, V -> f16 transposed [h][d][s] ================
__global__ __launch_bounds__(256)
void cvt_kv_kernel(const float* __restrict__ K, const float* __restrict__ V,
                   _Float16* __restrict__ K16, _Float16* __restrict__ V16T) {
  size_t t = ((size_t)blockIdx.x * 256 + threadIdx.x) * 4;   // flat over [h][s][d]
  f32x4 kv = *(const f32x4*)(K + t);
  v4h kh = { (_Float16)kv.x, (_Float16)kv.y, (_Float16)kv.z, (_Float16)kv.w };
  *(v4h*)(K16 + t) = kh;
  f32x4 vv = *(const f32x4*)(V + t);
  int d = (int)(t & 63), s = (int)((t >> 6) & (SEQ - 1)), h = (int)(t >> 18);
  _Float16* vb = V16T + (size_t)h * HD * SEQ + s;
  vb[(size_t)(d + 0) * SEQ] = (_Float16)vv.x;
  vb[(size_t)(d + 1) * SEQ] = (_Float16)vv.y;
  vb[(size_t)(d + 2) * SEQ] = (_Float16)vv.z;
  vb[(size_t)(d + 3) * SEQ] = (_Float16)vv.w;
}

// =================================== main attention kernel =================================
__global__ __launch_bounds__(128)
void sdpa_wmma_kernel(const float* __restrict__ Q, const uint32_t* __restrict__ BITS,
                      const _Float16* __restrict__ K16, const _Float16* __restrict__ V16T,
                      float* __restrict__ OutO, float* __restrict__ OutP) {
  __shared__ _Float16 Kst[2][CHK][64];    // double-buffered K chunk   [kk][d]
  __shared__ _Float16 Vt [2][64][CHK];    // double-buffered V^T chunk [d][kk]
  __shared__ _Float16 Pst[4][16][CHK];    // per-wave P tile           [qrow][kcol]

  const int tid  = threadIdx.x;
  const int lane = tid & 31;
  const int wave = tid >> 5;
  const int l16  = lane & 15;
  const int half = lane >> 4;
  const int h    = blockIdx.y;
  const int q0   = blockIdx.x * 64 + wave * 16;
  const int bshift = (q0 & 31) + half * 8;

  // ---- Q tile -> A operands; scale*log2e folded in so WMMA emits exp2-domain scores ----
  const float* Qrow = Q + ((size_t)h * SEQ + q0 + l16) * HD;
  v16h aq[2];
  #pragma unroll
  for (int c = 0; c < 2; ++c)
    #pragma unroll
    for (int p = 0; p < 8; ++p) {
      int d0 = c * 32 + half * 16 + 2 * p;
      aq[c][2 * p]     = (_Float16)(Qrow[d0] * SCL2);
      aq[c][2 * p + 1] = (_Float16)(Qrow[d0 + 1] * SCL2);
    }

  const _Float16* Kh = K16  + (size_t)h * SEQ * HD;
  const _Float16* Vh = V16T + (size_t)h * HD * SEQ;
  const uint32_t* Bq = BITS + (size_t)(q0 >> 5) * SEQ;

  float mx[8], sm[8];
  #pragma unroll
  for (int j = 0; j < 8; ++j) { mx[j] = -3.0e38f; sm[j] = 0.0f; }

  // ============ PASS 1: exact row max + row sum (TDM double-buffered K stream) ============
  if (wave == 0) tdm_load_2d((uint32_t)(uintptr_t)&Kst[0][0][0], Kh, 64, CHK, 64);
  for (int ic = 0; ic < NCH; ++ic) {
    const int buf = ic & 1;
    __syncthreads();                       // buffer buf^1 free for overwrite
    if (wave == 0) {
      if (ic + 1 < NCH) {
        tdm_load_2d((uint32_t)(uintptr_t)&Kst[buf ^ 1][0][0],
                    Kh + (size_t)(ic + 1) * CHK * HD, 64, CHK, 64);
        __builtin_amdgcn_s_wait_tensorcnt(1);   // chunk ic done; ic+1 stays in flight
      } else {
        __builtin_amdgcn_s_wait_tensorcnt(0);
      }
    }
    __syncthreads();                       // staged data visible to all waves
    const int kc = ic * CHK;
    #pragma unroll
    for (int n = 0; n < 4; ++n) {
      v8f c = {};
      #pragma unroll
      for (int cc = 0; cc < 2; ++cc)
        c = wmma32f16(aq[cc], ld16h(&Kst[buf][n * 16 + l16][cc * 32 + half * 16]), c);
      const int kcol = kc + n * 16 + l16;
      const uint32_t w = Bq[kcol] >> bshift;   // 16 query rows' mask bits in one word
      #pragma unroll
      for (int j = 0; j < 8; ++j) {
        float s = c[j];
        if (!((w >> j) & 1u)) s = NEG2;
        float nm = fmaxf(mx[j], s);
        sm[j] = sm[j] * exp2f(mx[j] - nm) + exp2f(s - nm);
        mx[j] = nm;
      }
    }
  }
  // 16-lane reduction (rows 0-7 in lanes 0-15, rows 8-15 in lanes 16-31)
  #pragma unroll
  for (int off = 1; off < 16; off <<= 1) {
    #pragma unroll
    for (int j = 0; j < 8; ++j) {
      float om = __shfl_xor(mx[j], off, 32);
      float os = __shfl_xor(sm[j], off, 32);
      float nm = fmaxf(mx[j], om);
      sm[j] = sm[j] * exp2f(mx[j] - nm) + os * exp2f(om - nm);
      mx[j] = nm;
    }
  }
  float b2[8];                               // softmax denominator folded into exponent
  #pragma unroll
  for (int j = 0; j < 8; ++j) b2[j] = mx[j] + log2f(sm[j]);

  v8f o[4];
  #pragma unroll
  for (int n = 0; n < 4; ++n) o[n] = (v8f){};

  float* Pout = OutP + (size_t)h * SEQ * SEQ;

  // ====== PASS 2: probs (NT 1GB stream) + O += P*V (TDM double-buffered K+V streams) ======
  __syncthreads();                           // pass-1 readers done before refilling buf0
  if (wave == 0) {
    tdm_load_2d((uint32_t)(uintptr_t)&Kst[0][0][0], Kh, 64, CHK, 64);
    tdm_load_2d((uint32_t)(uintptr_t)&Vt [0][0][0], Vh, CHK, 64, SEQ);
  }
  for (int ic = 0; ic < NCH; ++ic) {
    const int buf = ic & 1;
    __syncthreads();
    if (wave == 0) {
      if (ic + 1 < NCH) {
        tdm_load_2d((uint32_t)(uintptr_t)&Kst[buf ^ 1][0][0],
                    Kh + (size_t)(ic + 1) * CHK * HD, 64, CHK, 64);
        tdm_load_2d((uint32_t)(uintptr_t)&Vt[buf ^ 1][0][0],
                    Vh + (size_t)(ic + 1) * CHK, CHK, 64, SEQ);
        __builtin_amdgcn_s_wait_tensorcnt(2);
      } else {
        __builtin_amdgcn_s_wait_tensorcnt(0);
      }
    }
    __syncthreads();
    const int kc = ic * CHK;
    #pragma unroll
    for (int n = 0; n < 4; ++n) {
      v8f c = {};
      #pragma unroll
      for (int cc = 0; cc < 2; ++cc)
        c = wmma32f16(aq[cc], ld16h(&Kst[buf][n * 16 + l16][cc * 32 + half * 16]), c);
      const int kcol = kc + n * 16 + l16;
      const uint32_t w = Bq[kcol] >> bshift;
      #pragma unroll
      for (int j = 0; j < 8; ++j) {
        float s = c[j];
        if (!((w >> j) & 1u)) s = NEG2;
        float pw = exp2f(s - b2[j]);         // exp(s-m)/l with /l folded into exponent
        __builtin_nontemporal_store(pw, Pout + (size_t)(q0 + half * 8 + j) * SEQ + kcol);
        Pst[wave][half * 8 + j][n * 16 + l16] = (_Float16)pw;
      }
    }
    __syncthreads();
    #pragma unroll
    for (int cc2 = 0; cc2 < 2; ++cc2) {      // two 16x32 P sub-tiles -> 8 PV WMMAs
      v8h plo = *(const v8h*)&Pst[wave][l16][cc2 * 32 + half * 16];
      v8h phi = *(const v8h*)&Pst[wave][l16][cc2 * 32 + half * 16 + 8];
      v16h ap = __builtin_shufflevector(plo, phi, 0,1,2,3,4,5,6,7,8,9,10,11,12,13,14,15);
      #pragma unroll
      for (int n = 0; n < 4; ++n)
        o[n] = wmma32f16(ap, ld16h(&Vt[buf][n * 16 + l16][cc2 * 32 + half * 16]), o[n]);
    }
  }

  // ---- write O tile ----
  float* Og = OutO + ((size_t)h * SEQ + q0) * HD;
  #pragma unroll
  for (int n = 0; n < 4; ++n)
    #pragma unroll
    for (int j = 0; j < 8; ++j)
      Og[(size_t)(half * 8 + j) * HD + n * 16 + l16] = o[n][j];
}

extern "C" void kernel_launch(void* const* d_in, const int* in_sizes, int n_in,
                              void* d_out, int out_size, void* d_ws, size_t ws_size,
                              hipStream_t stream) {
  (void)in_sizes; (void)n_in; (void)out_size; (void)ws_size;
  const float* q    = (const float*)d_in[0];
  const float* k    = (const float*)d_in[1];
  const float* v    = (const float*)d_in[2];
  const int*   mask = (const int*)d_in[3];
  float* out_o = (float*)d_out;
  float* out_p = out_o + (size_t)NH * SEQ * HD;

  uint8_t* ws = (uint8_t*)d_ws;
  uint32_t*  bits = (uint32_t*)ws;                        // 2 MB
  _Float16*  K16  = (_Float16*)(ws + (size_t)(2  << 20)); // 8 MB
  _Float16*  V16T = (_Float16*)(ws + (size_t)(10 << 20)); // 8 MB

  pack_mask_kernel<<<dim3(SEQ / 256, SEQ / 32), 256, 0, stream>>>(mask, bits);
  cvt_kv_kernel<<<dim3((NH * SEQ * HD / 4) / 256), 256, 0, stream>>>(k, v, K16, V16T);
  sdpa_wmma_kernel<<<dim3(SEQ / 64, NH), 128, 0, stream>>>(q, bits, K16, V16T, out_o, out_p);
}
